// aformer_29918742184358
// MI455X (gfx1250) — compile-verified, matrix operand order checked by reading, not run on hardware
//
#include <hip/hip_runtime.h>
#include <hip/hip_bf16.h>
#include <math.h>
#include <stdint.h>

// ---------------------------------------------------------------------------
// Types for CDNA5 WMMA (wave32, 16x16x32 f16 -> f32)
// ---------------------------------------------------------------------------
typedef __attribute__((ext_vector_type(16))) _Float16 v16h;
typedef __attribute__((ext_vector_type(8)))  _Float16 v8h;
typedef __attribute__((ext_vector_type(8)))  float    v8f;

#define TM 128
#define TN 128
#define TK 32
#define LDSS 48   // halves per LDS row: 96B stride -> 8/16B aligned chunks, conflict-reducing

__device__ __forceinline__ float gelu_f(float x) {
  return 0.5f * x * (1.0f + erff(x * 0.70710678118654752f));
}
__device__ __forceinline__ float softplus_f(float x) {
  return (x > 20.f) ? x : log1pf(expf(x));
}

// ---------------------------------------------------------------------------
// Generic fused WMMA GEMM: C = epilogue( prologue(A) @ Bf16 + bias )
//  A: f32, staged through LDS with fused BN(scale/shift)+GELU im2col (conv) or
//     dense/transposed access (attention/FFN).
//  B: pre-cast f16 [Npad][Kpad], DMA'd to LDS via global_load_async_to_lds_b64.
//  prologue: 0 = dense A, 1 = im2col + BN + GELU, 2 = im2col raw
//  epilogue: 0 = none, 1 = GELU, 2 = relu(scale*x)+stab, 3 = +resid, 4 = *zscale
// ---------------------------------------------------------------------------
struct GemmParams {
  const float* A; const _Float16* Bt; const float* bias; const float* resid; const float* zscale;
  const float* bn_s; const float* bn_t;
  float* C;
  int M, N, K, Kpad;
  int lda, aOff, aTrans;
  int ldc, cOff;
  int prologue;
  int ksize, pad, cin, Lin;
  int epilogue;
  float scale, stab;
  int zStride, zOff;
};

__global__ __launch_bounds__(256) void wmma_gemm(GemmParams p) {
  __shared__ alignas(16) _Float16 Alds[TM * LDSS];
  __shared__ alignas(16) _Float16 Blds[TN * LDSS]; // [n][k]
  const int tid  = threadIdx.x;
  const int lane = tid & 31;
  const int wave = tid >> 5;
  const int m0 = blockIdx.x * TM;
  const int n0 = blockIdx.y * TN;
  const int wm = (wave >> 1) * 32;   // 4 wave-rows x 2 wave-cols
  const int wn = (wave & 1) * 64;
  v8f acc[2][4] = {};

  const int nK = (p.K + TK - 1) / TK;
  for (int kt = 0; kt < nK; ++kt) {
    const int k0 = kt * TK;
    if (kt + 1 < nK)
      __builtin_prefetch(p.Bt + (size_t)n0 * p.Kpad + (k0 + TK), 0, 1); // global_prefetch_b8

    // ---- B tile: async DMA global(f16) -> LDS, 4-half chunks per lane.
    // Kpad is a multiple of 32 and Npad a multiple of 128, so no bounds checks.
    for (int c = tid; c < (TN * TK) / 4; c += 256) {
      int n  = c >> 3;
      int kq = (c & 7) * 4;
      unsigned ldsOff = (unsigned)(uintptr_t)(const void*)&Blds[n * LDSS + kq];
      const _Float16* g = p.Bt + (size_t)(n0 + n) * p.Kpad + (k0 + kq);
      asm volatile("global_load_async_to_lds_b64 %0, %1, off"
                   :: "v"(ldsOff), "v"(g)
                   : "memory");
    }

    // ---- A tile: fused BN+GELU im2col (or dense cast), overlaps the async DMA
    for (int idx = tid; idx < TM * TK; idx += 256) {
      int r = idx >> 5, kk = idx & 31;
      int gi = m0 + r, gj = k0 + kk;
      float val = 0.f;
      if (gi < p.M && gj < p.K) {
        if (p.prologue == 0) {
          val = p.aTrans ? p.A[(size_t)gj * p.lda + p.aOff + gi]
                         : p.A[(size_t)gi * p.lda + p.aOff + gj];
        } else {
          int t  = gj / p.cin;
          int ic = gj - t * p.cin;
          int sr = gi + t - p.pad;
          if (sr >= 0 && sr < p.Lin) {
            float x = p.A[(size_t)sr * p.lda + p.aOff + ic];
            if (p.prologue == 1) x = gelu_f(x * p.bn_s[ic] + p.bn_t[ic]);
            val = x;
          }
        }
      }
      Alds[r * LDSS + kk] = (_Float16)val;
    }

    asm volatile("s_wait_asynccnt 0" ::: "memory"); // async LDS writes visible
    __syncthreads();

    // ---- fragments per ISA 16-bit layout:
    // lanes 0-15: K 0-7 (elts 0-7), K 16-23 (elts 8-15); lanes 16-31: K 8-15, 24-31
    const int lr   = lane & 15;
    const int kOff = (lane >> 4) * 8;
    v16h af[2], bf[4];
#pragma unroll
    for (int i = 0; i < 2; ++i) {
      const _Float16* ap = &Alds[(wm + i * 16 + lr) * LDSS + kOff];
      v8h alo = *(const v8h*)ap;
      v8h ahi = *(const v8h*)(ap + 16);
      af[i] = __builtin_shufflevector(alo, ahi, 0,1,2,3,4,5,6,7,8,9,10,11,12,13,14,15);
    }
#pragma unroll
    for (int j = 0; j < 4; ++j) {
      const _Float16* bp = &Blds[(wn + j * 16 + lr) * LDSS + kOff];
      v8h blo = *(const v8h*)bp;
      v8h bhi = *(const v8h*)(bp + 16);
      bf[j] = __builtin_shufflevector(blo, bhi, 0,1,2,3,4,5,6,7,8,9,10,11,12,13,14,15);
    }
#pragma unroll
    for (int i = 0; i < 2; ++i)
#pragma unroll
      for (int j = 0; j < 4; ++j)
        acc[i][j] = __builtin_amdgcn_wmma_f32_16x16x32_f16(
            false, af[i], false, bf[j], (short)0, acc[i][j], false, false);
    __syncthreads();
  }

  // ---- epilogue: C layout lanes 0-15 => M 0-7, lanes 16-31 => M 8-15, N = lane&15
  const int lr = lane & 15;
  const int mh = (lane >> 4) * 8;
#pragma unroll
  for (int i = 0; i < 2; ++i) {
#pragma unroll
    for (int j = 0; j < 4; ++j) {
      int col = n0 + wn + j * 16 + lr;
      if (col >= p.N) continue;
#pragma unroll
      for (int e = 0; e < 8; ++e) {
        int row = m0 + wm + i * 16 + mh + e;
        if (row >= p.M) continue;
        float v = acc[i][j][e];
        if (p.bias) v += p.bias[col];
        if      (p.epilogue == 1) v = gelu_f(v);
        else if (p.epilogue == 2) v = fmaxf(p.scale * v, 0.f) + p.stab;
        else if (p.epilogue == 3) v += p.resid[(size_t)row * p.ldc + p.cOff + col];
        else if (p.epilogue == 4) v *= p.zscale[(size_t)row * p.zStride + p.zOff];
        p.C[(size_t)row * p.ldc + p.cOff + col] = v;
      }
    }
  }
}

// ---------------------------------------------------------------------------
// Weight cast: f32 B(k,n) -> f16 Bt[n][Kpad], zero-padded (async-DMA friendly)
// trans=0: B(k,n)=src[k*ld+off+n]   trans=1: B(k,n)=src[n*ld+off+k]
// ---------------------------------------------------------------------------
__global__ void castB_kernel(const float* src, _Float16* dst, int K, int N,
                             int Kpad, int Npad, int ld, int off, int trans) {
  size_t i = (size_t)blockIdx.x * blockDim.x + threadIdx.x;
  if (i >= (size_t)Npad * Kpad) return;
  int n = (int)(i / Kpad), k = (int)(i - (size_t)n * Kpad);
  float v = 0.f;
  if (n < N && k < K)
    v = trans ? src[(size_t)n * ld + off + k] : src[(size_t)k * ld + off + n];
  dst[i] = (_Float16)v;
}

__global__ void bnprep_kernel(const float* g, const float* b, const float* m,
                              const float* v, float* s, float* t, int C) {
  int c = blockIdx.x * blockDim.x + threadIdx.x;
  if (c >= C) return;
  float sc = g[c] * rsqrtf(v[c] + 1e-5f);
  s[c] = sc;
  t[c] = b[c] - m[c] * sc;
}

// ---------------------------------------------------------------------------
// Light elementwise / reduction kernels
// ---------------------------------------------------------------------------
__global__ void maxpool_kernel(const float* in, float* out, int Lout, int C, int k) {
  int i = blockIdx.x * blockDim.x + threadIdx.x;
  if (i >= Lout * C) return;
  int l = i / C, c = i - l * C;
  float m = -INFINITY;
  for (int w = 0; w < k; ++w) m = fmaxf(m, in[(size_t)(l * k + w) * C + c]);
  out[i] = m;
}

__global__ void ln_kernel(const float* x, const float* g, const float* b, float* y, int C) {
  __shared__ float red[256];
  int row = blockIdx.x;
  const float* xr = x + (size_t)row * C;
  float s = 0.f, s2 = 0.f;
  for (int c = threadIdx.x; c < C; c += 256) { float v = xr[c]; s += v; s2 += v * v; }
  red[threadIdx.x] = s; __syncthreads();
  for (int o = 128; o > 0; o >>= 1) { if (threadIdx.x < o) red[threadIdx.x] += red[threadIdx.x + o]; __syncthreads(); }
  float mean = red[0] / C; __syncthreads();
  red[threadIdx.x] = s2; __syncthreads();
  for (int o = 128; o > 0; o >>= 1) { if (threadIdx.x < o) red[threadIdx.x] += red[threadIdx.x + o]; __syncthreads(); }
  float var = red[0] / C - mean * mean;
  float inv = rsqrtf(var + 1e-5f);
  for (int c = threadIdx.x; c < C; c += 256)
    y[(size_t)row * C + c] = (xr[c] - mean) * inv * g[c] + b[c];
}

__global__ void rotary_kernel(float* q, float* k, int L, int H, int Dh) {
  int half = Dh / 2;
  int i = blockIdx.x * blockDim.x + threadIdx.x;
  if (i >= L * H * half) return;
  int d = i % half; int t = i / half; int h = t % H; int l = t / H;
  float inv = powf(10000.f, -2.f * (float)d / (float)Dh);
  float ang = (float)l * inv;
  float s = sinf(ang), c = cosf(ang);
  size_t base = (size_t)l * (H * Dh) + (size_t)h * Dh + 2 * d;
  float q1 = q[base], q2 = q[base + 1];
  q[base] = q1 * c - q2 * s; q[base + 1] = q2 * c + q1 * s;
  float k1 = k[base], k2 = k[base + 1];
  k[base] = k1 * c - k2 * s; k[base + 1] = k2 * c + k1 * s;
}

__global__ void colsum_kernel(const float* kp, float* ks, int L, int HM) {
  int i = blockIdx.x * blockDim.x + threadIdx.x;
  if (i >= HM) return;
  float s = 0.f;
  for (int l = 0; l < L; ++l) s += kp[(size_t)l * HM + i];
  ks[i] = s;
}

__global__ void zinv_kernel(const float* qp, const float* ks, float* z, int L, int H, int Mf) {
  int i = blockIdx.x * blockDim.x + threadIdx.x;
  if (i >= L * H) return;
  int h = i % H, l = i / H;
  const float* qr = qp + (size_t)l * (H * Mf) + (size_t)h * Mf;
  const float* kr = ks + (size_t)h * Mf;
  float s = 0.f;
  for (int m = 0; m < Mf; ++m) s += qr[m] * kr[m];
  z[(size_t)l * H + h] = 1.f / s;
}

__global__ void tfrow_kernel(const float* tf, const float* W, const float* b, float* h, int T, int C) {
  int c = blockIdx.x * blockDim.x + threadIdx.x;
  if (c >= C) return;
  float s = b[c];
  for (int t = 0; t < T; ++t) s += tf[t] * W[(size_t)t * C + c];
  h[c] = gelu_f(s);
}

__global__ void concat_kernel(const float* x, const float* a, float* h, int L, int Cx, int Ca) {
  int C = Cx + Ca;
  int i = blockIdx.x * blockDim.x + threadIdx.x;
  if (i >= L * C) return;
  int l = i / C, c = i - l * C;
  float v = (c < Cx) ? x[(size_t)l * Cx + c] : a[(size_t)l * Ca + (c - Cx)];
  h[(size_t)(l + 1) * C + c] = v;
}

__global__ void head_kernel(const float* hf, const float* wr, const float* br,
                            const float* wa, const float* ba,
                            const float* assay, const float* aw, const float* ab,
                            float* out, int L, int C) {
  int l = blockIdx.x * blockDim.x + threadIdx.x;
  if (l >= L) return;
  float r = br[0], a = ba[0];
  for (int c = 0; c < C; ++c) { float v = hf[(size_t)l * C + c]; r += v * wr[c]; a += v * wa[c]; }
  float w = softplus_f(assay[0] * aw[0] + ab[0]);
  out[l] = w * softplus_f(r) + (1.f - w) * softplus_f(a);
}

// ---------------------------------------------------------------------------
// Host orchestration
// ---------------------------------------------------------------------------
struct WB { const _Float16* p; int Kp; };

static GemmParams mkG(int M, int N, int K, const float* A, int lda,
                      WB wb, const float* bias, float* C, int ldc) {
  GemmParams p{};
  p.M = M; p.N = N; p.K = K; p.Kpad = wb.Kp;
  p.A = A; p.lda = lda; p.Bt = wb.p; p.bias = bias; p.C = C; p.ldc = ldc;
  p.scale = 1.f;
  return p;
}
static void runG(const GemmParams& p, hipStream_t s) {
  dim3 g((p.M + TM - 1) / TM, (p.N + TN - 1) / TN);
  wmma_gemm<<<g, dim3(256), 0, s>>>(p);
}
static void runPool(const float* in, float* out, int Lout, int C, int k, hipStream_t s) {
  int n = Lout * C;
  maxpool_kernel<<<(n + 255) / 256, 256, 0, s>>>(in, out, Lout, C, k);
}

extern "C" void kernel_launch(void* const* d_in, const int* in_sizes, int n_in,
                              void* d_out, int out_size, void* d_ws, size_t ws_size,
                              hipStream_t stream) {
  (void)in_sizes; (void)n_in; (void)out_size; (void)ws_size;

  const int SEQL = 65536, ATACL = 16384;
  const int D = 1600, L = 513, H = 4, DH = 400, NBRF = 256, FF = 3200;
  const int FL = 448, FC = 128;
  const int FSEQ[6] = {768, 896, 1024, 1152, 1280, 1536};

  const float* seq   = (const float*)d_in[0];
  const float* atac  = (const float*)d_in[1];
  const float* tf    = (const float*)d_in[2];
  const float* assay = (const float*)d_in[3];

  // params flattened jax-pytree style (alphabetical dict keys), starting at 4
  int ix = 4;
  const int I_assay_b = ix++; const int I_assay_w = ix++;
  const int I_fln_b = ix++;   const int I_fln_g = ix++;
  const int I_fp_bn_b = ix++; const int I_fp_bn_g = ix++; const int I_fp_bn_m = ix++; const int I_fp_bn_v = ix++;
  const int I_fp_cb = ix++;   const int I_fp_cw = ix++;
  const int I_ha_b = ix++;    const int I_ha_w = ix++;
  const int I_hr_b = ix++;    const int I_hr_w = ix++;
  const int I_layers = ix;    ix += 6 * 14; // fc1_b,fc1_w,fc2_b,fc2_w,ln1_b,ln1_g,ln2_b,ln2_g,proj,wk,wo_b,wo_w,wq,wv
  const int I_sc_b = ix++;    const int I_sc_w = ix++;
  const int I_sca_b = ix++;   const int I_sca_w = ix++;
  const int I_sr_bn_b = ix++; const int I_sr_bn_g = ix++; const int I_sr_bn_m = ix++; const int I_sr_bn_v = ix++;
  const int I_sr_cb = ix++;   const int I_sr_cw = ix++;
  const int I_sra_bn_b = ix++; const int I_sra_bn_g = ix++; const int I_sra_bn_m = ix++; const int I_sra_bn_v = ix++;
  const int I_sra_cb = ix++;  const int I_sra_cw = ix++;
  const int I_tf_b = ix++;    const int I_tf_w = ix++;
  const int I_tower = ix;     ix += 36; // 6 x {bn_b,bn_g,bn_m,bn_v,conv_b,conv_w}
  const int I_ta = ix;        ix += 12; // 2 x same

  auto F = [&](int i) -> const float* { return (const float*)d_in[i]; };

  // workspace bump allocator
  size_t off = 0;
  auto allocB = [&](size_t bytes) -> void* {
    void* p = (char*)d_ws + off;
    off += (bytes + 255) & ~(size_t)255;
    return p;
  };
  auto alloc  = [&](size_t elems) -> float*     { return (float*)allocB(elems * 4); };
  auto allocH = [&](size_t elems) -> _Float16*  { return (_Float16*)allocB(elems * 2); };

  // f32 activation ping-pong + transformer scratch
  float* bufA = alloc((size_t)SEQL * 768);
  float* bufB = alloc((size_t)SEQL * 768);
  float* atA  = alloc((size_t)ATACL * 32);
  float* atB  = alloc((size_t)ATACL * 32);
  float* hbuf = alloc((size_t)L * D);
  float* lnb  = alloc((size_t)L * D);
  float* qb   = alloc((size_t)L * D);
  float* kb   = alloc((size_t)L * D);
  float* vb   = alloc((size_t)L * D);
  float* qpb  = alloc((size_t)L * H * NBRF);
  float* kpb  = alloc((size_t)L * H * NBRF);
  float* ksb  = alloc((size_t)H * NBRF);
  float* zb   = alloc((size_t)L * H);
  float* kvb  = alloc((size_t)H * NBRF * DH);
  float* ob   = alloc((size_t)L * D);
  float* ffnb = alloc((size_t)L * FF);
  float* hfin = alloc((size_t)FL * FC);

  // weight cast helper: f32 [K][N] (or [N][K] if trans) -> f16 [Npad][Kpad]
  auto castW = [&](const float* src, int K, int N, int ld, int offc, int trans) -> WB {
    int Kp = (K + 31) & ~31;
    int Np = (N + TN - 1) & ~(TN - 1);
    _Float16* d = allocH((size_t)Np * Kp);
    size_t n = (size_t)Np * Kp;
    castB_kernel<<<(unsigned)((n + 255) / 256), 256, 0, stream>>>(src, d, K, N, Kp, Np, ld, offc, trans);
    return WB{d, Kp};
  };
  // BN scale/shift prep
  auto bnprep = [&](int ig, int ib, int im, int iv, int C, float** S, float** T) {
    *S = alloc(C); *T = alloc(C);
    bnprep_kernel<<<(C + 255) / 256, 256, 0, stream>>>(F(ig), F(ib), F(im), F(iv), *S, *T, C);
  };

  // ---- one-time (per launch) weight casts to async-DMA-friendly f16 layout
  WB w_sc  = castW(F(I_sc_w), 60, 768, 768, 0, 0);
  WB w_sr  = castW(F(I_sr_cw), 768, 768, 768, 0, 0);
  WB w_tw[6];
  { int cin = 768;
    for (int t = 0; t < 6; ++t) { w_tw[t] = castW(F(I_tower + 6 * t + 5), 5 * cin, FSEQ[t], FSEQ[t], 0, 0); cin = FSEQ[t]; } }
  WB w_sca = castW(F(I_sca_w), 50, 32, 32, 0, 0);
  WB w_sra = castW(F(I_sra_cw), 32, 32, 32, 0, 0);
  WB w_ta0 = castW(F(I_ta + 5), 160, 32, 32, 0, 0);
  WB w_ta1 = castW(F(I_ta + 11), 160, 64, 64, 0, 0);
  WB w_fp  = castW(F(I_fp_cw), D, FC, FC, 0, 0);
  WB w_q[6], w_k[6], w_v[6], w_o[6], w_f1[6], w_f2[6], w_pj[6];
  for (int ly = 0; ly < 6; ++ly) {
    const int B0 = I_layers + ly * 14;
    w_f1[ly] = castW(F(B0 + 1), D, FF, FF, 0, 0);
    w_f2[ly] = castW(F(B0 + 3), FF, D, D, 0, 0);
    w_pj[ly] = castW(F(B0 + 8), DH, NBRF, DH, 0, 1); // proj is [N=256][K=400]
    w_k[ly]  = castW(F(B0 + 9), D, D, D, 0, 0);
    w_o[ly]  = castW(F(B0 + 11), D, D, D, 0, 0);
    w_q[ly]  = castW(F(B0 + 12), D, D, D, 0, 0);
    w_v[ly]  = castW(F(B0 + 13), D, D, D, 0, 0);
  }
  // dynamic-operand f16 buffers (re-cast each layer)
  int vKp = (L + 31) & ~31;     // 544
  _Float16* vbT[4]; _Float16* kvT[4];
  for (int hh = 0; hh < 4; ++hh) {
    vbT[hh] = allocH((size_t)512 * vKp);
    kvT[hh] = allocH((size_t)512 * 256);
  }

  float *S_sr, *T_sr, *S_sra, *T_sra, *S_fp, *T_fp;
  float *S_tw[6], *T_tw[6], *S_ta[2], *T_ta[2];
  bnprep(I_sr_bn_g, I_sr_bn_b, I_sr_bn_m, I_sr_bn_v, 768, &S_sr, &T_sr);
  bnprep(I_sra_bn_g, I_sra_bn_b, I_sra_bn_m, I_sra_bn_v, 32, &S_sra, &T_sra);
  bnprep(I_fp_bn_g, I_fp_bn_b, I_fp_bn_m, I_fp_bn_v, D, &S_fp, &T_fp);
  { int cin = 768;
    for (int t = 0; t < 6; ++t) {
      bnprep(I_tower + 6 * t + 1, I_tower + 6 * t + 0, I_tower + 6 * t + 2, I_tower + 6 * t + 3, cin, &S_tw[t], &T_tw[t]);
      cin = FSEQ[t];
    } }
  bnprep(I_ta + 1, I_ta + 0, I_ta + 2, I_ta + 3, 32, &S_ta[0], &T_ta[0]);
  bnprep(I_ta + 7, I_ta + 6, I_ta + 8, I_ta + 9, 32, &S_ta[1], &T_ta[1]);

  // ---------------- sequence path ----------------
  { GemmParams p = mkG(SEQL, 768, 60, seq, 4, w_sc, F(I_sc_b), bufA, 768);
    p.prologue = 2; p.ksize = 15; p.pad = 7; p.cin = 4; p.Lin = SEQL;
    runG(p, stream); }
  { GemmParams p = mkG(SEQL, 768, 768, bufA, 768, w_sr, F(I_sr_cb), bufB, 768);
    p.prologue = 1; p.ksize = 1; p.pad = 0; p.cin = 768; p.Lin = SEQL;
    p.bn_s = S_sr; p.bn_t = T_sr;
    p.epilogue = 3; p.resid = bufA;
    runG(p, stream); }
  runPool(bufB, bufA, SEQL / 2, 768, 2, stream);

  int Lc = SEQL / 2, cin = 768;
  for (int t = 0; t < 6; ++t) {
    int cout = FSEQ[t];
    GemmParams p = mkG(Lc, cout, 5 * cin, bufA, cin, w_tw[t], F(I_tower + 6 * t + 4), bufB, cout);
    p.prologue = 1; p.ksize = 5; p.pad = 2; p.cin = cin; p.Lin = Lc;
    p.bn_s = S_tw[t]; p.bn_t = T_tw[t];
    runG(p, stream);
    runPool(bufB, bufA, Lc / 2, cout, 2, stream);
    Lc /= 2; cin = cout;
  } // bufA: 512 x 1536

  // ---------------- atac path ----------------
  { GemmParams p = mkG(ATACL, 32, 50, atac, 1, w_sca, F(I_sca_b), atA, 32);
    p.prologue = 2; p.ksize = 50; p.pad = 24; p.cin = 1; p.Lin = ATACL;
    runG(p, stream); }
  { GemmParams p = mkG(ATACL, 32, 32, atA, 32, w_sra, F(I_sra_cb), atB, 32);
    p.prologue = 1; p.ksize = 1; p.pad = 0; p.cin = 32; p.Lin = ATACL;
    p.bn_s = S_sra; p.bn_t = T_sra;
    p.epilogue = 3; p.resid = atA;
    runG(p, stream); }
  runPool(atB, atA, 8192, 32, 2, stream);
  { GemmParams p = mkG(8192, 32, 160, atA, 32, w_ta0, F(I_ta + 4), atB, 32);
    p.prologue = 1; p.ksize = 5; p.pad = 2; p.cin = 32; p.Lin = 8192;
    p.bn_s = S_ta[0]; p.bn_t = T_ta[0];
    runG(p, stream); }
  runPool(atB, atA, 2048, 32, 4, stream);
  { GemmParams p = mkG(2048, 64, 160, atA, 32, w_ta1, F(I_ta + 10), atB, 64);
    p.prologue = 1; p.ksize = 5; p.pad = 2; p.cin = 32; p.Lin = 2048;
    p.bn_s = S_ta[1]; p.bn_t = T_ta[1];
    runG(p, stream); }
  runPool(atB, atA, 512, 64, 4, stream); // atA: 512 x 64

  // ---------------- token assembly ----------------
  tfrow_kernel<<<(D + 255) / 256, 256, 0, stream>>>(tf, F(I_tf_w), F(I_tf_b), hbuf, 1629, D);
  concat_kernel<<<(512 * D + 255) / 256, 256, 0, stream>>>(bufA, atA, hbuf, 512, 1536, 64);

  // ---------------- transformer (FAVOR+) ----------------
  for (int ly = 0; ly < 6; ++ly) {
    const int B0 = I_layers + ly * 14;
    const float *fc1_b = F(B0 + 0), *fc2_b = F(B0 + 2);
    const float *ln1_b = F(B0 + 4), *ln1_g = F(B0 + 5);
    const float *ln2_b = F(B0 + 6), *ln2_g = F(B0 + 7);
    const float *wo_b  = F(B0 + 10);

    ln_kernel<<<L, 256, 0, stream>>>(hbuf, ln1_g, ln1_b, lnb, D);
    runG(mkG(L, D, D, lnb, D, w_q[ly], nullptr, qb, D), stream);
    runG(mkG(L, D, D, lnb, D, w_k[ly], nullptr, kb, D), stream);
    runG(mkG(L, D, D, lnb, D, w_v[ly], nullptr, vb, D), stream);
    rotary_kernel<<<(L * H * (DH / 2) + 255) / 256, 256, 0, stream>>>(qb, kb, L, H, DH);

    for (int hh = 0; hh < H; ++hh) { // qp/kp = relu(q_rot @ proj^T / 16) + stab
      GemmParams p = mkG(L, NBRF, DH, qb, D, w_pj[ly], nullptr, qpb, H * NBRF);
      p.aOff = hh * DH; p.cOff = hh * NBRF;
      p.epilogue = 2; p.scale = 0.0625f; p.stab = 1e-3f;
      runG(p, stream);
      p.A = kb; p.C = kpb;
      runG(p, stream);
    }
    colsum_kernel<<<(H * NBRF + 255) / 256, 256, 0, stream>>>(kpb, ksb, L, H * NBRF);
    for (int hh = 0; hh < H; ++hh) { // v_h -> f16 [400pad][Lpad] for async path
      size_t n = (size_t)512 * vKp;
      castB_kernel<<<(unsigned)((n + 255) / 256), 256, 0, stream>>>(vb, vbT[hh], L, DH, vKp, 512, D, hh * DH, 0);
      GemmParams p = mkG(NBRF, DH, L, kpb, H * NBRF, WB{vbT[hh], vKp}, nullptr, kvb + (size_t)hh * NBRF * DH, DH);
      p.aTrans = 1; p.aOff = hh * NBRF;
      runG(p, stream);
    }
    zinv_kernel<<<(L * H + 255) / 256, 256, 0, stream>>>(qpb, ksb, zb, L, H, NBRF);
    for (int hh = 0; hh < H; ++hh) { // o_h = (qp_h @ kv_h) * z
      size_t n = (size_t)512 * 256;
      castB_kernel<<<(unsigned)((n + 255) / 256), 256, 0, stream>>>(kvb + (size_t)hh * NBRF * DH, kvT[hh],
                                                                    NBRF, DH, 256, 512, DH, 0, 0);
      GemmParams p = mkG(L, DH, NBRF, qpb, H * NBRF, WB{kvT[hh], 256}, nullptr, ob, D);
      p.aOff = hh * NBRF; p.cOff = hh * DH;
      p.epilogue = 4; p.zscale = zb; p.zStride = H; p.zOff = hh;
      runG(p, stream);
    }
    { GemmParams p = mkG(L, D, D, ob, D, w_o[ly], wo_b, hbuf, D);
      p.epilogue = 3; p.resid = hbuf; runG(p, stream); }

    ln_kernel<<<L, 256, 0, stream>>>(hbuf, ln2_g, ln2_b, lnb, D);
    { GemmParams p = mkG(L, FF, D, lnb, D, w_f1[ly], fc1_b, ffnb, FF);
      p.epilogue = 1; runG(p, stream); }
    { GemmParams p = mkG(L, D, FF, ffnb, FF, w_f2[ly], fc2_b, hbuf, D);
      p.epilogue = 3; p.resid = hbuf; runG(p, stream); }
  }

  // ---------------- final head ----------------
  ln_kernel<<<L, 256, 0, stream>>>(hbuf, F(I_fln_g), F(I_fln_b), lnb, D);
  { GemmParams p = mkG(FL, FC, D, lnb + 33 * D, D, w_fp, F(I_fp_cb), hfin, FC);
    p.prologue = 1; p.ksize = 1; p.pad = 0; p.cin = D; p.Lin = FL;
    p.bn_s = S_fp; p.bn_t = T_fp;
    p.epilogue = 1;
    runG(p, stream); }
  head_kernel<<<(FL + 255) / 256, 256, 0, stream>>>(
      hfin, F(I_hr_w), F(I_hr_b), F(I_ha_w), F(I_ha_b),
      assay, F(I_assay_w), F(I_assay_b), (float*)d_out, FL, FC);
}
